// GraphCNN_34643206209863
// MI455X (gfx1250) — matrix operand order repaired and checked by reading, not compiled, not run on hardware
//
#include <hip/hip_runtime.h>
#include <hip/hip_bf16.h>

typedef __attribute__((ext_vector_type(16))) _Float16 v16h;
typedef __attribute__((ext_vector_type(8)))  _Float16 v8h;
typedef __attribute__((ext_vector_type(4)))  _Float16 v4h;
typedef __attribute__((ext_vector_type(8)))  float    v8f;
typedef __attribute__((ext_vector_type(4)))  float    v4f;

#define NN 8192
#define FF 128
#define LDS_STRIDE 40   // halves; 80B row pitch -> 16B aligned segments, conflict-free

static __device__ __forceinline__ v16h cat16(v8h lo, v8h hi) {
  v16h r;
#pragma unroll
  for (int i = 0; i < 8; ++i) { r[i] = lo[i]; r[8 + i] = hi[i]; }
  return r;
}

static __device__ __forceinline__ v8f wmma_f16(v16h a, v16h b, v8f c) {
  // 8 args: (neg_a, A, neg_b, B, c_mod, C, reuse_a, reuse_b)
  return __builtin_amdgcn_wmma_f32_16x16x32_f16(false, a, false, b, (short)0, c,
                                                false, false);
}

// Row-normalize f32 [N,128] -> f16 normalized (Qn) and f16 raw values (Vh).
__global__ __launch_bounds__(128) void prep_rows(const float* __restrict__ X,
                                                 _Float16* __restrict__ Qn,
                                                 _Float16* __restrict__ Vh) {
  const int row = blockIdx.x;
  const int t   = threadIdx.x;
  float v  = X[(size_t)row * FF + t];
  float ss = v * v;
#pragma unroll
  for (int off = 16; off > 0; off >>= 1) ss += __shfl_xor(ss, off, 32);
  __shared__ float ws[4];
  if ((t & 31) == 0) ws[t >> 5] = ss;
  __syncthreads();
  float tot = ws[0] + ws[1] + ws[2] + ws[3];
  float nrm = fmaxf(sqrtf(tot), 1e-12f);
  float rn  = 1.0f / nrm;
  Qn[(size_t)row * FF + t] = (_Float16)(v * rn);
  Vh[(size_t)row * FF + t] = (_Float16)v;
}

// One fused AGNN layer: Out[i,f] = leaky( softmax_j( mask(a) beta*cos(i,j) ) @ V )
// 4 waves/WG, each wave owns a 16-row tile; loop over 32-column blocks.
__global__ __launch_bounds__(128) void agnn_layer(
    const _Float16* __restrict__ Qn,   // [N,128] normalized, f16
    const _Float16* __restrict__ Vh,   // [N,128] values, f16
    const float* __restrict__ Adj,     // [N,N] mask, f32
    const float* __restrict__ beta,    // scalar
    float* __restrict__ Out)           // [N,128] f32 (leaky applied)
{
  __shared__ _Float16 VT[FF * LDS_STRIDE];   // V^T tile: VT[f][j], j in [0,32)

  const int t    = threadIdx.x;
  const int wave = t >> 5;
  const int lane = t & 31;
  const int li   = lane & 15;     // i / f / j within-16 index
  const int g    = lane >> 4;     // half-wave group
  const int ib   = (blockIdx.x * 4 + wave) * 16;   // this wave's row base
  const float bl = beta[0] * 1.44269504088896f;    // beta * log2(e): exp(b*s)=exp2(bl*s)

  // Preload B fragments for matmul 1: rows ib+li of Qn, K-chunks of 32.
  // B layout (16-bit, 32x16): lane holds column n=li, half e<8 -> K = 8g + e ; e>=8 -> K = 16 + 8g + (e-8).
  v16h Bq[4];
  {
    const _Float16* qrow = Qn + (size_t)(ib + li) * FF;
#pragma unroll
    for (int c = 0; c < 4; ++c)
      Bq[c] = cat16(*(const v8h*)(qrow + 32 * c + 8 * g), *(const v8h*)(qrow + 32 * c + 16 + 8 * g));   // 2x16B, 16B-aligned
  }

  v8f O[8];
#pragma unroll
  for (int c = 0; c < 8; ++c)
#pragma unroll
    for (int e = 0; e < 8; ++e) O[c][e] = 0.0f;
  float lsum = 0.0f;

  const float* arow = Adj + (size_t)(ib + li) * NN;

  for (int jb = 0; jb < NN; jb += 32) {
    // Prefetch the adjacency stream ~8 iterations (1KB/lane) ahead.
    if (jb + 256 < NN) __builtin_prefetch(arow + jb + 256, 0, 1);

    // ---- stage V^T tile in LDS (cooperative, transposing write) ----
    __syncthreads();
    {
      const int j0 = (t & 7) * 4;       // 4 consecutive j
      const int f0 = (t >> 3) * 8;      // 8 consecutive f
      v8h r0 = *(const v8h*)(Vh + (size_t)(jb + j0 + 0) * FF + f0);
      v8h r1 = *(const v8h*)(Vh + (size_t)(jb + j0 + 1) * FF + f0);
      v8h r2 = *(const v8h*)(Vh + (size_t)(jb + j0 + 2) * FF + f0);
      v8h r3 = *(const v8h*)(Vh + (size_t)(jb + j0 + 3) * FF + f0);
#pragma unroll
      for (int f = 0; f < 8; ++f) {
        v4h pk; pk[0] = r0[f]; pk[1] = r1[f]; pk[2] = r2[f]; pk[3] = r3[f];
        *(v4h*)&VT[(f0 + f) * LDS_STRIDE + j0] = pk;     // ds_store_b64
      }
    }
    __syncthreads();

    // ---- matmul 1: S^T tiles, D[m=j][n=i], K=128 ----
    // A layout: lane holds row m=li, half e<8 -> k=32c+8g+e ; e>=8 -> k=32c+16+8g+(e-8)
    v8f S0, S1;
#pragma unroll
    for (int e = 0; e < 8; ++e) { S0[e] = 0.0f; S1[e] = 0.0f; }
    const _Float16* qj0 = Qn + (size_t)(jb + li) * FF;
    const _Float16* qj1 = Qn + (size_t)(jb + 16 + li) * FF;
#pragma unroll
    for (int c = 0; c < 4; ++c) {
      v16h a0 = cat16(*(const v8h*)(qj0 + 32 * c + 8 * g),
                      *(const v8h*)(qj0 + 32 * c + 16 + 8 * g));
      S0 = wmma_f16(a0, Bq[c], S0);
      v16h a1 = cat16(*(const v8h*)(qj1 + 32 * c + 8 * g),
                      *(const v8h*)(qj1 + 32 * c + 16 + 8 * g));
      S1 = wmma_f16(a1, Bq[c], S1);
    }

    // ---- mask + exp (no max-subtraction: beta*cos is bounded) ----
    // D layout: lane holds i=li; reg r of tile0 -> j = 8g+r, tile1 -> j = 16+8g+r
    v4f m0a = *(const v4f*)(arow + jb + 8 * g);
    v4f m0b = *(const v4f*)(arow + jb + 8 * g + 4);
    v4f m1a = *(const v4f*)(arow + jb + 16 + 8 * g);
    v4f m1b = *(const v4f*)(arow + jb + 16 + 8 * g + 4);

    float p0[8], p1[8];
#pragma unroll
    for (int r = 0; r < 8; ++r) {
      float mv0 = (r < 4) ? m0a[r] : m0b[r - 4];
      float mv1 = (r < 4) ? m1a[r] : m1b[r - 4];
      p0[r] = (mv0 > 0.0f) ? exp2f(bl * S0[r]) : 0.0f;
      p1[r] = (mv1 > 0.0f) ? exp2f(bl * S1[r]) : 0.0f;
      lsum += p0[r] + p1[r];
    }

    // ---- repack P into A-fragment of O = P @ V (pure in-lane) ----
    // A half e<8 -> k=j=8g+e  (tile0 reg e); e>=8 -> k=16+8g+(e-8) (tile1 reg e-8)
    v16h Ap;
#pragma unroll
    for (int e = 0; e < 8; ++e) {
      Ap[e]     = (_Float16)p0[e];
      Ap[8 + e] = (_Float16)p1[e];
    }

    // ---- matmul 2: O[chunk] += P @ V, B from LDS V^T ----
    // B half e<8 -> K=j=8g+e ; e>=8 -> K=j=16+8g+(e-8) : VT[f = 16*ch + li][...]
#pragma unroll
    for (int ch = 0; ch < 8; ++ch) {
      const _Float16* vt = &VT[(16 * ch + li) * LDS_STRIDE];
      v16h Bv = cat16(*(const v8h*)(vt + 8 * g), *(const v8h*)(vt + 16 + 8 * g));
      O[ch] = wmma_f16(Ap, Bv, O[ch]);
    }
  }

  // ---- epilogue: normalize by row sum, leaky-relu, store ----
  float ltot = lsum + __shfl_xor(lsum, 16, 32);   // full row sum for i = li
  float rinv = 1.0f / ltot;
#pragma unroll
  for (int r = 0; r < 8; ++r) {
    float ri = __shfl(rinv, 8 * g + r, 32);       // bcast 1/l for row i = 8g+r
#pragma unroll
    for (int ch = 0; ch < 8; ++ch) {
      float h = O[ch][r] * ri;                    // D: lane n=f=li(+16ch), reg m=i=8g+r
      h = (h > 0.0f) ? h : 0.3f * h;
      Out[(size_t)(ib + 8 * g + r) * FF + 16 * ch + li] = h;
    }
  }
}

extern "C" void kernel_launch(void* const* d_in, const int* in_sizes, int n_in,
                              void* d_out, int out_size, void* d_ws, size_t ws_size,
                              hipStream_t stream) {
  const float* x  = (const float*)d_in[0];   // [8192,128]
  const float* a  = (const float*)d_in[1];   // [8192,8192]
  const float* b1 = (const float*)d_in[2];   // [1]
  const float* b2 = (const float*)d_in[3];   // [1]
  float* out = (float*)d_out;

  char* ws = (char*)d_ws;
  _Float16* Qn = (_Float16*)ws;                               // 2 MB
  _Float16* Vh = (_Float16*)(ws + (size_t)2 * 1024 * 1024);   // 2 MB
  float*    H1 = (float*)   (ws + (size_t)4 * 1024 * 1024);   // 4 MB

  // Layer 1
  prep_rows<<<NN, 128, 0, stream>>>(x, Qn, Vh);
  agnn_layer<<<NN / 64, 128, 0, stream>>>(Qn, Vh, a, b1, H1);
  // Layer 2
  prep_rows<<<NN, 128, 0, stream>>>(H1, Qn, Vh);
  agnn_layer<<<NN / 64, 128, 0, stream>>>(Qn, Vh, a, b2, out);
}
